// Summariser_11922829214291
// MI455X (gfx1250) — matrix operand order repaired and checked by reading, not compiled, not run on hardware
//
#include <hip/hip_runtime.h>
#include <hip/hip_bf16.h>
#include <math.h>
#include <cstdint>

// ---------------- problem constants (from reference) ----------------
constexpr int B_ = 2, S_ = 4096, H_ = 768, HEADS_ = 12, DH_ = 64;
constexpr int W_ = 256, FF_ = 3072, NL_ = 2, NC_ = S_ / W_;

typedef __bf16 bf16;
typedef __attribute__((ext_vector_type(16))) __bf16 v16bf;
typedef __attribute__((ext_vector_type(8)))  float  v8f;

__device__ __forceinline__ v8f wmma_bf16(v16bf a, v16bf b, v8f c) {
  // D(f32 16x16) = A(bf16 16x32) * B(bf16 32x16) + C
  return __builtin_amdgcn_wmma_f32_16x16x32_bf16(
      false, a, false, b, (short)0, c, false, false);
}

// ---------------- gfx1250 async global->LDS copies (ASYNCcnt) ----------------
// The clang builtin exists but its pointer parameters use HIP-internal
// address-space types with no source spelling; inline asm (doc-sanctioned
// route) bypasses the frontend type check and is toolchain-portable.
// LDS address comes from the vdst VGPR; low 32 bits of a flat shared pointer are
// the LDS byte address (aperture rule), so a truncated cast is correct.
__device__ __forceinline__ void async_copy16(const bf16* g, bf16* l) {
  asm volatile("global_load_async_to_lds_b128 %0, %1, off"
               :
               : "v"((uint32_t)(uintptr_t)l), "v"((uint64_t)(uintptr_t)g)
               : "memory");
}
__device__ __forceinline__ void async_wait_all() {
  asm volatile("s_wait_asynccnt 0x0" ::: "memory");
}

// ---------------- cross-lane helpers via ds_swizzle (wave32) ----------------
template <int OFF>
__device__ __forceinline__ float swzf(float x) {
  return __int_as_float(__builtin_amdgcn_ds_swizzle(__float_as_int(x), OFF));
}
__device__ __forceinline__ float rowmax16(float x) {
  x = fmaxf(x, swzf<0x041F>(x));
  x = fmaxf(x, swzf<0x081F>(x));
  x = fmaxf(x, swzf<0x101F>(x));
  x = fmaxf(x, swzf<0x201F>(x));
  return x;
}
__device__ __forceinline__ float rowsum16(float x) {
  x += swzf<0x041F>(x);
  x += swzf<0x081F>(x);
  x += swzf<0x101F>(x);
  x += swzf<0x201F>(x);
  return x;
}
__device__ __forceinline__ float wavesum32(float x) {
  x += swzf<0x041F>(x);
  x += swzf<0x081F>(x);
  x += swzf<0x101F>(x);
  x += swzf<0x201F>(x);
  x += swzf<0x401F>(x);
  return x;
}

__device__ __forceinline__ float gelu_exact(float x) {
  return 0.5f * x * (1.0f + erff(x * 0.70710678118654752f));
}

// ---------------- fp32 -> bf16 conversion (weights) ----------------
__global__ __launch_bounds__(256) void cvt_bf16_kernel(const float* __restrict__ in,
                                                       bf16* __restrict__ out, int n) {
  int idx = blockIdx.x * 1024 + threadIdx.x;
#pragma unroll
  for (int i = 0; i < 4; ++i) {
    int j = idx + i * 256;
    if (j < n) out[j] = (bf16)in[j];
  }
}

// ---------------- embedding + LayerNorm (fp32 + bf16 outputs) ----------------
__global__ __launch_bounds__(256) void embed_ln_kernel(
    const float* __restrict__ fv, const float* __restrict__ pos,
    const float* __restrict__ te, const float* __restrict__ g,
    const float* __restrict__ bb, float* __restrict__ outf,
    bf16* __restrict__ outh) {
  const int row  = blockIdx.x;           // 0 .. B*S-1
  const int spos = row & (S_ - 1);
  const int tid  = threadIdx.x;
  const size_t off = (size_t)row * H_;
  float vals[3], s = 0.f, ss = 0.f;
#pragma unroll
  for (int i = 0; i < 3; ++i) {
    int col = tid + i * 256;
    float v = fv[off + col] + pos[(size_t)spos * H_ + col] + te[col];
    vals[i] = v; s += v; ss += v * v;
  }
  s = wavesum32(s); ss = wavesum32(ss);
  __shared__ float rs[8], rss[8];
  if ((tid & 31) == 0) { rs[tid >> 5] = s; rss[tid >> 5] = ss; }
  __syncthreads();
  float ts = 0.f, tss = 0.f;
#pragma unroll
  for (int i = 0; i < 8; ++i) { ts += rs[i]; tss += rss[i]; }
  float mu = ts * (1.0f / H_);
  float var = tss * (1.0f / H_) - mu * mu;
  float rstd = rsqrtf(var + 1e-5f);
#pragma unroll
  for (int i = 0; i < 3; ++i) {
    int col = tid + i * 256;
    float o = (vals[i] - mu) * rstd * g[col] + bb[col];
    outf[off + col] = o;
    outh[off + col] = (bf16)o;
  }
}

// ---------------- residual + LayerNorm (fp32 + bf16 outputs) ----------------
__global__ __launch_bounds__(256) void ln_residual_kernel(
    const float* __restrict__ base, const float* __restrict__ delta,
    const float* __restrict__ g, const float* __restrict__ bb,
    float* __restrict__ outf, bf16* __restrict__ outh) {
  const int row = blockIdx.x;
  const int tid = threadIdx.x;
  const size_t off = (size_t)row * H_;
  float vals[3], s = 0.f, ss = 0.f;
#pragma unroll
  for (int i = 0; i < 3; ++i) {
    int col = tid + i * 256;
    float v = base[off + col] + delta[off + col];
    vals[i] = v; s += v; ss += v * v;
  }
  s = wavesum32(s); ss = wavesum32(ss);
  __shared__ float rs[8], rss[8];
  if ((tid & 31) == 0) { rs[tid >> 5] = s; rss[tid >> 5] = ss; }
  __syncthreads();
  float ts = 0.f, tss = 0.f;
#pragma unroll
  for (int i = 0; i < 8; ++i) { ts += rs[i]; tss += rss[i]; }
  float mu = ts * (1.0f / H_);
  float var = tss * (1.0f / H_) - mu * mu;
  float rstd = rsqrtf(var + 1e-5f);
#pragma unroll
  for (int i = 0; i < 3; ++i) {
    int col = tid + i * 256;
    float o = (vals[i] - mu) * rstd * g[col] + bb[col];
    outf[off + col] = o;
    outh[off + col] = (bf16)o;
  }
}

// ---------------- bf16 WMMA GEMM with async double-buffered LDS ----------------
// A: [M,K] bf16, W: [K,N] bf16 (both row-major, 16B-aligned rows).
// Macro tile 128(M) x 64(N), K-step 32; 8 waves (4Mx2N), each wave 32x32
// -> 4 v_wmma per K-step with 2x fragment reuse.
template <int ACT, int OBF>  // ACT: 0 none, 1 exact GELU; OBF: fp32 vs bf16 out
__global__ __launch_bounds__(256) void gemm_kernel(
    const bf16* __restrict__ A, const bf16* __restrict__ Wt,
    const float* __restrict__ bias, float* __restrict__ Cf,
    bf16* __restrict__ Ch, int M, int N, int K, float scale) {
  __shared__ bf16 As[2][128][40];  // row stride 80B (16B multiple)
  __shared__ bf16 Bs[2][32][72];   // row stride 144B (16B multiple)
  const int tid  = threadIdx.x;
  const int lane = tid & 31;
  const int wave = tid >> 5;
  const int wm   = wave >> 1;      // 0..3 -> 32-row slice
  const int wn   = wave & 1;       // 0..1 -> 32-col slice
  const int hf   = lane >> 4;
  const int l16  = lane & 15;
  const int bm   = blockIdx.y * 128;
  const int bn   = blockIdx.x * 64;

  auto stage = [&](int kt, int buf) {
    const int k0 = kt * 32;
    // A tile 128x32 bf16 = 512 x 16B chunks (2/thread)
#pragma unroll
    for (int i = 0; i < 2; ++i) {
      int cid = tid + i * 256;
      int r = cid >> 2, cg = (cid & 3) * 8;
      async_copy16(&A[(size_t)(bm + r) * K + k0 + cg], &As[buf][r][cg]);
    }
    // B tile 32x64 bf16 = 256 x 16B chunks (1/thread)
    int r = tid >> 3, cg = (tid & 7) * 8;
    async_copy16(&Wt[(size_t)(k0 + r) * N + bn + cg], &Bs[buf][r][cg]);
  };

  v8f z = {};
  v8f acc[2][2] = {{z, z}, {z, z}};

  const int nk = K >> 5;
  stage(0, 0);
  async_wait_all();
  __syncthreads();

  for (int kt = 0; kt < nk; ++kt) {
    const int cur = kt & 1;
    if (kt + 1 < nk) stage(kt + 1, cur ^ 1);  // overlap with WMMA on cur

    v16bf am[2], bw[2];
#pragma unroll
    for (int s2 = 0; s2 < 2; ++s2) {
      // A fragment: lane row = l16; halves split K per CDNA5 layout
#pragma unroll
      for (int e = 0; e < 16; ++e) {
        int vv = e >> 1, j = e & 1;
        int kk = (vv < 4 ? 0 : 16) + (hf ? 8 : 0) + ((vv & 3) << 1) + j;
        am[s2][e] = As[cur][wm * 32 + s2 * 16 + l16][kk];
      }
      // B fragment: N across lanes; K = (e<8 ? 0 : 16) + hf*8 + (e&7)
#pragma unroll
      for (int e = 0; e < 16; ++e)
        bw[s2][e] = Bs[cur][(e < 8 ? 0 : 16) + hf * 8 + (e & 7)][wn * 32 + s2 * 16 + l16];
    }
#pragma unroll
    for (int i = 0; i < 2; ++i)
#pragma unroll
      for (int j = 0; j < 2; ++j)
        acc[i][j] = wmma_bf16(am[i], bw[j], acc[i][j]);

    async_wait_all();   // next tile landed (every wave waits its own loads)
    __syncthreads();    // all waves done with cur + all loads visible
  }

  // epilogue: C layout VGPR r, lane l -> row r+8*hf, col l16
#pragma unroll
  for (int i = 0; i < 2; ++i)
#pragma unroll
    for (int j = 0; j < 2; ++j)
#pragma unroll
      for (int r = 0; r < 8; ++r) {
        int row = bm + wm * 32 + i * 16 + r + 8 * hf;
        int col = bn + wn * 32 + j * 16 + l16;
        float v = acc[i][j][r] + bias[col];
        if (ACT == 1) v = gelu_exact(v);
        v *= scale;
        if (OBF) Ch[(size_t)row * N + col] = (bf16)v;
        else     Cf[(size_t)row * N + col] = v;
      }
}

// ---------------- sliding-window flash attention (bf16 in/out) ----------------
// One block per (b, head, 256-query chunk). Band mask |kg - qg| <= W.
// S, W multiples of 32 => every non-skipped 32-key block is fully in range.
__global__ __launch_bounds__(256) void swattn_kernel(
    const bf16* __restrict__ Qg, const bf16* __restrict__ Kg,
    const bf16* __restrict__ Vg, bf16* __restrict__ Og) {
  const int c  = blockIdx.x % NC_;
  const int h  = (blockIdx.x / NC_) % HEADS_;
  const int b  = blockIdx.x / (NC_ * HEADS_);
  const int tid  = threadIdx.x;
  const int lane = tid & 31;
  const int wave = tid >> 5;
  const int hf   = lane >> 4;
  const int l16  = lane & 15;

  __shared__ bf16 Qs[256][72];   // 144B row stride (16B multiple)
  __shared__ bf16 Ks[32][72];
  __shared__ bf16 Vs[32][72];
  __shared__ bf16 Ps[8][16][36]; // per-wave P re-layout buffer

  const size_t headcol = (size_t)h * DH_;
  const size_t rowbase = (size_t)b * S_;

  // stage the 256x64 query chunk (pre-scaled by 1/sqrt(dh)) via async b128
#pragma unroll
  for (int i = 0; i < 8; ++i) {
    int cid = tid + i * 256;
    int r = cid >> 3, cg = (cid & 7) * 8;
    async_copy16(&Qg[(rowbase + c * W_ + r) * H_ + headcol + cg], &Qs[r][cg]);
  }
  async_wait_all();
  __syncthreads();

  // prebuild Q A-fragments: 2 query tiles x 2 K-steps (dh 0-31 / 32-63)
  v16bf aq[2][2];
#pragma unroll
  for (int t = 0; t < 2; ++t)
#pragma unroll
    for (int dsp = 0; dsp < 2; ++dsp)
#pragma unroll
      for (int e = 0; e < 16; ++e) {
        int vv = e >> 1, j = e & 1;
        int kk = (vv < 4 ? 0 : 16) + (hf ? 8 : 0) + ((vv & 3) << 1) + j;
        aq[t][dsp][e] = Qs[wave * 32 + t * 16 + l16][dsp * 32 + kk];
      }

  float mrow[2][8], lrow[2][8];
  v8f oacc[2][4];
  {
    v8f z = {};
#pragma unroll
    for (int t = 0; t < 2; ++t) {
#pragma unroll
      for (int r = 0; r < 8; ++r) { mrow[t][r] = -1e30f; lrow[t][r] = 0.f; }
#pragma unroll
      for (int n = 0; n < 4; ++n) oacc[t][n] = z;
    }
  }

  const int kwin0 = (c - 1) * W_;  // start of 3-chunk key window
  for (int kb = 0; kb < 24; ++kb) {
    const int ks0 = kwin0 + kb * 32;
    if (ks0 < 0 || ks0 >= S_) continue;  // uniform across block; blocks never partial

    // stage K,V 32x64 blocks via async b128 (1 chunk each per thread)
    {
      int r = tid >> 3, cg = (tid & 7) * 8;
      size_t goff = (rowbase + ks0 + r) * H_ + headcol + cg;
      async_copy16(&Kg[goff], &Ks[r][cg]);
      async_copy16(&Vg[goff], &Vs[r][cg]);
    }
    async_wait_all();
    __syncthreads();

#pragma unroll
    for (int t = 0; t < 2; ++t) {
      // scores S = Q*K^T : 16x32 via two 16x16 WMMA tiles (K-dim = dh)
      v8f sc[2];
#pragma unroll
      for (int ns = 0; ns < 2; ++ns) {
        v8f cs = {};
#pragma unroll
        for (int dsp = 0; dsp < 2; ++dsp) {
          v16bf bk;
#pragma unroll
          for (int e = 0; e < 16; ++e)
            bk[e] = Ks[ns * 16 + l16][dsp * 32 + (e < 8 ? 0 : 16) + hf * 8 + (e & 7)];
          cs = wmma_bf16(aq[t][dsp], bk, cs);
        }
        sc[ns] = cs;
      }

      // band mask + online softmax (rows live across 16-lane halves)
      const int qbase = c * W_ + wave * 32 + t * 16;
#pragma unroll
      for (int r = 0; r < 8; ++r) {
        int qg  = qbase + r + 8 * hf;
        float s0 = sc[0][r], s1 = sc[1][r];
        int da = ks0 + l16 - qg;      if (da < 0) da = -da;
        int db = ks0 + 16 + l16 - qg; if (db < 0) db = -db;
        if (da > W_) s0 = -1e30f;
        if (db > W_) s1 = -1e30f;
        float mx   = rowmax16(fmaxf(s0, s1));
        float mnew = fmaxf(mrow[t][r], mx);
        float p0 = (s0 < -1e29f) ? 0.f : __expf(s0 - mnew);
        float p1 = (s1 < -1e29f) ? 0.f : __expf(s1 - mnew);
        float rsum = rowsum16(p0 + p1);
        float fac  = __expf(mrow[t][r] - mnew);
        lrow[t][r] = lrow[t][r] * fac + rsum;
        mrow[t][r] = mnew;
#pragma unroll
        for (int n = 0; n < 4; ++n) oacc[t][n][r] *= fac;
        // write P (C layout) to per-wave LDS buffer for A-fragment re-layout
        Ps[wave][r + 8 * hf][l16]      = (bf16)p0;
        Ps[wave][r + 8 * hf][16 + l16] = (bf16)p1;
      }
      // per-wave LDS ops are in-order: no block barrier needed here

      // O += P(16x32) * V(32x64)
      v16bf ap;
#pragma unroll
      for (int e = 0; e < 16; ++e) {
        int vv = e >> 1, j = e & 1;
        int kk = (vv < 4 ? 0 : 16) + (hf ? 8 : 0) + ((vv & 3) << 1) + j;
        ap[e] = Ps[wave][l16][kk];
      }
#pragma unroll
      for (int n = 0; n < 4; ++n) {
        v16bf bv;
#pragma unroll
        for (int e = 0; e < 16; ++e)
          bv[e] = Vs[(e < 8 ? 0 : 16) + hf * 8 + (e & 7)][n * 16 + l16];
        oacc[t][n] = wmma_bf16(ap, bv, oacc[t][n]);
      }
    }
    __syncthreads();  // before K/V overwrite next block
  }

  // normalize and write out bf16 (merged-head layout [B*S, H])
#pragma unroll
  for (int t = 0; t < 2; ++t)
#pragma unroll
    for (int r = 0; r < 8; ++r) {
      int qg = c * W_ + wave * 32 + t * 16 + r + 8 * hf;
      float inv = 1.0f / lrow[t][r];
#pragma unroll
      for (int n = 0; n < 4; ++n)
        Og[(rowbase + qg) * H_ + headcol + n * 16 + l16] =
            (bf16)(oacc[t][n][r] * inv);
    }
}

// ---------------- pooled first-token output ----------------
__global__ __launch_bounds__(256) void pooled_kernel(const float* __restrict__ x,
                                                     float* __restrict__ out) {
  int i = blockIdx.x * 256 + threadIdx.x;
  if (i < B_ * H_) {
    int b = i / H_, col = i % H_;
    out[i] = x[(size_t)b * S_ * H_ + col];
  }
}

// ---------------- host orchestration ----------------
extern "C" void kernel_launch(void* const* d_in, const int* in_sizes, int n_in,
                              void* d_out, int out_size, void* d_ws, size_t ws_size,
                              hipStream_t stream) {
  (void)in_sizes; (void)n_in; (void)out_size; (void)ws_size;
  const float* full_vec = (const float*)d_in[0];
  const float* pos_emb  = (const float*)d_in[1];
  const float* type_emb = (const float*)d_in[2];
  const float* emb_g    = (const float*)d_in[3];
  const float* emb_b    = (const float*)d_in[4];
  // per-layer params start at index 5, 16 tensors each:
  // wq,bq,wk,bk,wv,bv,wo,bo,ln1_g,ln1_b,wi,bi,wf,bf,ln2_g,ln2_b

  const int M = B_ * S_;                 // 8192 rows
  const size_t SZ = (size_t)M * H_;

  char* wsp = (char*)d_ws;
  auto alloc = [&](size_t bytes) -> void* {
    void* r = (void*)wsp;
    wsp += (bytes + 255) & ~(size_t)255;
    return r;
  };
  float* x0f = (float*)alloc(SZ * 4);
  float* x1f = (float*)alloc(SZ * 4);
  float* t0f = (float*)alloc(SZ * 4);
  bf16*  x0h = (bf16*)alloc(SZ * 2);
  bf16*  x1h = (bf16*)alloc(SZ * 2);
  bf16*  qh  = (bf16*)alloc(SZ * 2);
  bf16*  kh  = (bf16*)alloc(SZ * 2);
  bf16*  vh  = (bf16*)alloc(SZ * 2);
  bf16*  ah  = (bf16*)alloc(SZ * 2);
  bf16*  hh  = (bf16*)alloc((size_t)M * FF_ * 2);

  // bf16 weight copies: wq,wk,wv,wo,wi,wf per layer
  const size_t wsz[6]  = {(size_t)H_ * H_, (size_t)H_ * H_, (size_t)H_ * H_,
                          (size_t)H_ * H_, (size_t)H_ * FF_, (size_t)FF_ * H_};
  const int    wofs[6] = {0, 2, 4, 6, 10, 12};
  bf16* wh[NL_][6];
  for (int l = 0; l < NL_; ++l)
    for (int j = 0; j < 6; ++j) wh[l][j] = (bf16*)alloc(wsz[j] * 2);

  const dim3 blk(256);
  const dim3 ln_grid(M);
  const dim3 gemm_h(H_ / 64, M / 128);    // N=768 -> 12 x 64
  const dim3 gemm_ff(FF_ / 64, M / 128);  // N=3072 -> 48 x 64
  const dim3 attn_grid(B_ * HEADS_ * NC_);

  // weight conversions (deterministic, redone each call)
  for (int l = 0; l < NL_; ++l)
    for (int j = 0; j < 6; ++j) {
      const float* src = (const float*)d_in[5 + l * 16 + wofs[j]];
      int n = (int)wsz[j];
      cvt_bf16_kernel<<<dim3((n + 1023) / 1024), blk, 0, stream>>>(src, wh[l][j], n);
    }

  // embeddings + LN
  embed_ln_kernel<<<ln_grid, blk, 0, stream>>>(full_vec, pos_emb, type_emb,
                                               emb_g, emb_b, x0f, x0h);

  const float qscale = 0.125f;  // 1/sqrt(64)
  for (int l = 0; l < NL_; ++l) {
    const float* const* P = (const float* const*)&d_in[5 + l * 16];
    const float *bq = P[1], *bkk = P[3], *bv = P[5], *bo = P[7];
    const float *g1 = P[8], *b1 = P[9], *bi = P[11], *bf = P[13];
    const float *g2 = P[14], *b2 = P[15];
    bf16 *whq = wh[l][0], *whk = wh[l][1], *whv = wh[l][2];
    bf16 *who = wh[l][3], *whi = wh[l][4], *whf = wh[l][5];

    gemm_kernel<0, 1><<<gemm_h, blk, 0, stream>>>(x0h, whq, bq, nullptr, qh,
                                                  M, H_, H_, qscale);
    gemm_kernel<0, 1><<<gemm_h, blk, 0, stream>>>(x0h, whk, bkk, nullptr, kh,
                                                  M, H_, H_, 1.0f);
    gemm_kernel<0, 1><<<gemm_h, blk, 0, stream>>>(x0h, whv, bv, nullptr, vh,
                                                  M, H_, H_, 1.0f);
    swattn_kernel<<<attn_grid, blk, 0, stream>>>(qh, kh, vh, ah);
    gemm_kernel<0, 0><<<gemm_h, blk, 0, stream>>>(ah, who, bo, t0f, nullptr,
                                                  M, H_, H_, 1.0f);
    ln_residual_kernel<<<ln_grid, blk, 0, stream>>>(x0f, t0f, g1, b1, x1f, x1h);
    gemm_kernel<1, 1><<<gemm_ff, blk, 0, stream>>>(x1h, whi, bi, nullptr, hh,
                                                   M, FF_, H_, 1.0f);
    gemm_kernel<0, 0><<<gemm_h, blk, 0, stream>>>(hh, whf, bf, t0f, nullptr,
                                                  M, H_, FF_, 1.0f);
    ln_residual_kernel<<<ln_grid, blk, 0, stream>>>(x1f, t0f, g2, b2, x0f, x0h);
  }

  pooled_kernel<<<dim3((B_ * H_ + 255) / 256), blk, 0, stream>>>(x0f, (float*)d_out);
}
